// conv_lstm_26225070310102
// MI455X (gfx1250) — matrix operand order repaired and checked by reading, not compiled, not run on hardware
//
#include <hip/hip_runtime.h>
#include <hip/hip_bf16.h>

// ---------------- problem constants ----------------
#define B_SZ   64
#define T_SZ   1024
#define D_INC  128
#define NB     256
#define KW     5
#define L_SZ   510          // (1024-5)/2+1
#define HID    512
#define G4     2048         // 4*HID
#define KC     640          // D_INC*KW  (conv implicit-GEMM K)
#define KR     768          // NB + HID  (fused recurrence K)
#define EPSBN  1e-5f

// LDS partition (elements of bf16) for the persistent kernel
#define SW_ELEMS   (128 * KR)          // weight slice: 128 gate cols x 768
#define SSEQ_OFF   (SW_ELEMS)          // seq panel: 16 x 256
#define SH_OFF     (SW_ELEMS + 16 * NB)// h panel:  16 x 512
#define SMEM_ELEMS (SW_ELEMS + 16 * NB + 16 * HID)   // 110592 elems = 221184 B

static_assert((B_SZ * L_SZ) % 16 == 0, "M tiles");
static_assert(KC % 32 == 0 && KR % 32 == 0, "K tiles");

// ---------------- types for WMMA ----------------
typedef __attribute__((ext_vector_type(16))) __bf16 bf16x16;
typedef __attribute__((ext_vector_type(8)))  __bf16 bf16x8;
typedef __attribute__((ext_vector_type(8)))  float  f32x8;
typedef __attribute__((ext_vector_type(4)))  unsigned int u32x4;
typedef __attribute__((ext_vector_type(8)))  unsigned int u32x8;

union FragU {
    bf16x16 v;
    bf16x8  h[2];
    __bf16  e[16];
};

__device__ __forceinline__ __bf16 f2bf(float f) {
    union { float f; unsigned u; } v; v.f = f;
    unsigned r = v.u + 0x7FFFu + ((v.u >> 16) & 1u);   // round-to-nearest-even
    union { unsigned short s; __bf16 b; } o;
    o.s = (unsigned short)(r >> 16);
    return o.b;
}

__device__ __forceinline__ float sigmoidf(float x) {
    return 1.0f / (1.0f + __expf(-x));
}

// ---- Tensor Data Mover: 2D tile load, 2-descriptor-group form ----
// D# packing per CDNA5 ISA. data_size = 2 bytes.
// Dims/strides in elements, addresses in bytes. Tracked by TENSORcnt.
__device__ __forceinline__ void tdm_load_2d(unsigned lds_off_bytes,
                                            const void* gptr,
                                            unsigned tile_d0, unsigned tile_d1,
                                            unsigned long long tdim0,
                                            unsigned long long tdim1,
                                            unsigned long long stride0)
{
    unsigned long long ga = (unsigned long long)(uintptr_t)gptr;
    u32x4 g0;
    g0[0] = 1u;                                            // count=1, user mode
    g0[1] = lds_off_bytes;                                 // lds_addr
    g0[2] = (unsigned)(ga & 0xffffffffu);                  // global_addr[31:0]
    g0[3] = (unsigned)((ga >> 32) & 0x01ffffffu) | (2u << 30); // addr[56:32], type=2

    unsigned long long q0 = (1ull << 16)                   // data_size = 1 -> 2 bytes
                          | ((tdim0 & 0xffffull) << 48);   // tensor_dim0[15:0] @48
    unsigned long long q1 = ((tdim0 >> 16) & 0xffffull)    // tensor_dim0[31:16] @64
                          | ((tdim1 & 0xffffffffull) << 16)        // tensor_dim1 @80
                          | ((unsigned long long)(tile_d0 & 0xffffu) << 48); // tile_dim0 @112
    unsigned long long q2 = (unsigned long long)(tile_d1 & 0xffffu)          // tile_dim1 @128
                          | ((stride0 & 0xffffffffull) << 32);     // dim0_stride[31:0] @160
    unsigned long long q3 = ((stride0 >> 32) & 0xffffull);         // dim0_stride[47:32] @192

    u32x8 g1;
    g1[0] = (unsigned)q0;  g1[1] = (unsigned)(q0 >> 32);
    g1[2] = (unsigned)q1;  g1[3] = (unsigned)(q1 >> 32);
    g1[4] = (unsigned)q2;  g1[5] = (unsigned)(q2 >> 32);
    g1[6] = (unsigned)q3;  g1[7] = (unsigned)(q3 >> 32);

    asm volatile("tensor_load_to_lds %0, %1" :: "s"(g0), "s"(g1) : "memory");
}

// Grid-wide sense barrier in workspace memory: bar[0]=count, bar[1]=generation.
__device__ __forceinline__ void gridBarrier(unsigned* bar, unsigned nblk) {
    __syncthreads();
    if (threadIdx.x == 0) {
        __threadfence();
        volatile unsigned* gen = bar + 1;
        const unsigned g = *gen;
        if (atomicAdd(bar, 1u) == nblk - 1u) {
            bar[0] = 0u;
            __threadfence();
            atomicAdd((unsigned*)(bar + 1), 1u);
        } else {
            while (*gen == g) { __builtin_amdgcn_s_sleep(2); }
        }
        __threadfence();
    }
    __syncthreads();
}

// ---------------- prep kernels ----------------
__global__ void prep_xbf(const float* __restrict__ x, __bf16* __restrict__ xbf) {
    int i = blockIdx.x * blockDim.x + threadIdx.x;
    if (i < B_SZ * T_SZ * D_INC) xbf[i] = f2bf(x[i]);
}

// conv_w [NB][D_INC][KW] -> wconv [NB][kk*128+d] (bf16): each WMMA K-window of
// 32 lies inside one tap kk (128 % 32 == 0) and is contiguous in d.
__global__ void prep_conv_w(const float* __restrict__ cw, __bf16* __restrict__ wconv) {
    int idx = blockIdx.x * blockDim.x + threadIdx.x;
    if (idx >= NB * KC) return;
    int n = idx / KC, r = idx - n * KC;
    int kk = r >> 7, d = r & 127;
    wconv[idx] = f2bf(cw[((size_t)n * D_INC + d) * KW + kk]);
}

// wcomb [G4][KR] = [w_ih | w_hh] (bf16), row-major with K contiguous
__global__ void prep_comb_w(const float* __restrict__ wih, const float* __restrict__ whh,
                            __bf16* __restrict__ wcomb) {
    int idx = blockIdx.x * blockDim.x + threadIdx.x;
    if (idx >= G4 * KR) return;
    int g = idx / KR, c = idx - g * KR;
    float v = (c < NB) ? wih[(size_t)g * NB + c] : whh[(size_t)g * HID + (c - NB)];
    wcomb[idx] = f2bf(v);
}

__global__ void prep_misc(const float* __restrict__ bih, const float* __restrict__ bhh,
                          const float* __restrict__ gamma, const float* __restrict__ beta,
                          const float* __restrict__ mean,  const float* __restrict__ var,
                          float* bcomb, float* bnscale, float* bnshift,
                          float* hbuf, float* cbuf, __bf16* hbf,
                          float* utbuf, float* ubin, float* dotpart, unsigned* bar) {
    int tid = blockIdx.x * blockDim.x + threadIdx.x;    // 32768 threads
    if (tid < G4) bcomb[tid] = bih[tid] + bhh[tid];
    if (tid < NB) {
        float s = gamma[tid] * rsqrtf(var[tid] + EPSBN);
        bnscale[tid] = s;
        bnshift[tid] = beta[tid] - mean[tid] * s;
    }
    if (tid < B_SZ * HID) { hbuf[tid] = 0.f; cbuf[tid] = 0.f; hbf[tid] = f2bf(0.f); }
    if (tid < B_SZ) { utbuf[tid] = 1.f; ubin[tid] = 1.f; }     // u0 = 1
    if (tid < B_SZ * 8) dotpart[tid] = 0.f;
    if (tid < 2) bar[tid] = 0u;
}

// ---------------- Conv1d + ReLU + BN as implicit WMMA GEMM ----------------
// M = B*L (tile per block), N = NB (2 n-tiles per wave x 8 waves), K = 640.
// A rows are contiguous 640-element bf16 runs of x (K order kk*128+d).
__global__ void __launch_bounds__(256)
conv_bn_wmma(const __bf16* __restrict__ xbf,        // [B][T][D_INC] bf16
             const __bf16* __restrict__ wconv,      // [NB][KC] bf16 (reordered)
             const float* __restrict__ convb,
             const float* __restrict__ bnscale,
             const float* __restrict__ bnshift,
             __bf16* __restrict__ seqbf)            // [B*L][NB] bf16
{
    const int lane   = threadIdx.x & 31;
    const int wave   = threadIdx.x >> 5;
    const int laneLo = lane & 15;
    const int laneHi = lane >> 4;

    const int mTile = blockIdx.x;                    // 0..2039
    const int m  = mTile * 16 + laneLo;              // A-row for this lane
    const int b  = m / L_SZ;
    const int l  = m - b * L_SZ;
    const __bf16* xrow = xbf + ((size_t)b * T_SZ + 2 * l) * D_INC;  // 640 contiguous K

    for (int nt = 0; nt < 2; ++nt) {
        const int nTile = wave * 2 + nt;
        const int n = nTile * 16 + laneLo;           // B-col / output col for this lane
        const __bf16* wrow = wconv + (size_t)n * KC;

        f32x8 acc = {0.f, 0.f, 0.f, 0.f, 0.f, 0.f, 0.f, 0.f};
        #pragma unroll 5
        for (int kb = 0; kb < KC / 32; ++kb) {
            const int kBase = kb * 32;
            FragU a, bf;
            const __bf16* pa = xrow + kBase + 8 * laneHi;    // A: halves0-7=K+0..7, 8-15=+16
            a.h[0] = *(const bf16x8*)(pa);
            a.h[1] = *(const bf16x8*)(pa + 16);
            const __bf16* pb = wrow + kBase + 8 * laneHi;    // B: halves0-7=K+0..7, 8-15=+16
            bf.h[0] = *(const bf16x8*)(pb);
            bf.h[1] = *(const bf16x8*)(pb + 16);
            acc = __builtin_amdgcn_wmma_f32_16x16x32_bf16(
                false, a.v, false, bf.v, (short)0, acc, false, false);
        }
        // epilogue: bias + relu + folded batchnorm, store bf16
        const float cb = convb[n], sc = bnscale[n], sh = bnshift[n];
        #pragma unroll
        for (int j = 0; j < 8; ++j) {
            const int mr = mTile * 16 + 8 * laneHi + j;
            float y = acc[j] + cb;
            y = fmaxf(y, 0.f);
            y = y * sc + sh;
            seqbf[(size_t)mr * NB + n] = f2bf(y);
        }
    }
}

// ---------------- persistent SkipLSTM (fused x_proj + recurrence) ----------------
// 64 blocks x 256 thr = 512 waves == 512 gate tiles of [64 x 2048].
// Block's 192KB weight slice lives in LDS for all 510 steps (TDM-loaded once).
// Per step the TDM stages the shared A panel (seq_t 8KB + h 16KB) into LDS.
extern __shared__ __bf16 smem[];

__global__ void __launch_bounds__(256)
skiplstm_persistent(const __bf16* __restrict__ seqbf,   // [B*L][NB]
                    const __bf16* __restrict__ wcomb,   // [G4][KR]
                    const float* __restrict__ bcomb,    // [G4]
                    const float* __restrict__ wu,       // [HID]
                    const float* __restrict__ bu,       // [1]
                    float* __restrict__ gates,          // [B][G4]
                    float* __restrict__ hbuf,           // [B][HID]
                    float* __restrict__ cbuf,           // [B][HID]
                    __bf16* __restrict__ hbf,           // [B][HID]
                    float* __restrict__ utbuf,          // [B]
                    float* __restrict__ ubin,           // [B]  round(ut)
                    float* __restrict__ dotpart,        // [B][8]
                    unsigned* __restrict__ bar,
                    float* __restrict__ out)            // [B][HID]
{
    const int lane   = threadIdx.x & 31;
    const int laneLo = lane & 15;
    const int laneHi = lane >> 4;
    const int wave   = threadIdx.x >> 5;

    const int mTile   = blockIdx.x >> 4;          // 0..3 (shared by all 8 waves)
    const int bBase   = mTile * 16;               // batch rows of this block's A panel
    const int colBase = (blockIdx.x & 15) * 128;  // this block's 128 gate columns
    const int nCol    = colBase + wave * 16 + laneLo;
    const float bias  = bcomb[nCol];

    const __bf16* sW   = smem;                    // [128][KR] weight slice
    const __bf16* sSeq = smem + SSEQ_OFF;         // [16][NB]
    const __bf16* sH   = smem + SH_OFF;           // [16][HID]
    const __bf16* sWrow = sW + (size_t)(wave * 16 + laneLo) * KR;

    const unsigned ldsW   = (unsigned)(uintptr_t)&smem[0];
    const unsigned ldsSeq = (unsigned)(uintptr_t)&smem[SSEQ_OFF];
    const unsigned ldsH   = (unsigned)(uintptr_t)&smem[SH_OFF];

    const int gtid = blockIdx.x * 256 + threadIdx.x;  // 0..16383
    const int e0   = gtid * 2;                        // two consecutive elements
    const int b2   = e0 >> 9;
    const int j0   = e0 & 511;

    // ---- one-time: TDM-load this block's weight slice (128 x 768 bf16) ----
    if (wave == 0) {
        tdm_load_2d(ldsW, wcomb + (size_t)colBase * KR,
                    /*tile*/ KR, 128, /*tensor*/ KR, 128, /*stride0*/ KR);
        __builtin_amdgcn_s_wait_tensorcnt(0);
    }
    __syncthreads();

    for (int t = 0; t < L_SZ; ++t) {
        // ---- stage A panel for this step via the Tensor Data Mover ----
        if (wave == 0) {
            tdm_load_2d(ldsSeq, seqbf + ((size_t)bBase * L_SZ + t) * NB,
                        /*tile*/ NB, 16, /*tensor*/ NB, 16,
                        /*stride0*/ (unsigned long long)L_SZ * NB);
            tdm_load_2d(ldsH, hbf + (size_t)bBase * HID,
                        /*tile*/ 16 * HID, 1, /*tensor*/ 16 * HID, 1,
                        /*stride0*/ 16 * HID);
            __builtin_amdgcn_s_wait_tensorcnt(0);
        }
        __syncthreads();

        // ---- Phase 1: gates = [seq_t | h] x wcomb^T + bias (LDS-resident) ----
        f32x8 acc = {0.f, 0.f, 0.f, 0.f, 0.f, 0.f, 0.f, 0.f};
        const __bf16* aSeq = sSeq + (size_t)laneLo * NB;
        const __bf16* aH   = sH + (size_t)laneLo * HID;
        #pragma unroll 4
        for (int kb = 0; kb < NB / 32; ++kb) {          // K 0..255 from seq_t
            const int kBase = kb * 32;
            FragU a, bf;
            const __bf16* pa = aSeq + kBase + 8 * laneHi;
            a.h[0] = *(const bf16x8*)(pa);
            a.h[1] = *(const bf16x8*)(pa + 16);
            const __bf16* pb = sWrow + kBase + 8 * laneHi;
            bf.h[0] = *(const bf16x8*)(pb);
            bf.h[1] = *(const bf16x8*)(pb + 16);
            acc = __builtin_amdgcn_wmma_f32_16x16x32_bf16(
                false, a.v, false, bf.v, (short)0, acc, false, false);
        }
        #pragma unroll 4
        for (int kb = 0; kb < HID / 32; ++kb) {         // K 256..767 from h
            const int kBase = kb * 32;
            FragU a, bf;
            const __bf16* pa = aH + kBase + 8 * laneHi;
            a.h[0] = *(const bf16x8*)(pa);
            a.h[1] = *(const bf16x8*)(pa + 16);
            const __bf16* pb = sWrow + NB + kBase + 8 * laneHi;
            bf.h[0] = *(const bf16x8*)(pb);
            bf.h[1] = *(const bf16x8*)(pb + 16);
            acc = __builtin_amdgcn_wmma_f32_16x16x32_bf16(
                false, a.v, false, bf.v, (short)0, acc, false, false);
        }
        #pragma unroll
        for (int j = 0; j < 8; ++j) {
            const int mr = bBase + 8 * laneHi + j;
            gates[(size_t)mr * G4 + nCol] = acc[j] + bias;
        }
        gridBarrier(bar, gridDim.x);

        // ---- Phase 2: elementwise LSTM update + wave-local partial dot ----
        {
            const float u  = ubin[b2];
            const float um = 1.f - u;
            float pd = 0.f;
            #pragma unroll
            for (int q = 0; q < 2; ++q) {
                const int j = j0 + q;
                const size_t gb = (size_t)b2 * G4 + j;
                const float ii = sigmoidf(gates[gb]);
                const float ff = sigmoidf(gates[gb + 512]);
                const float gg = tanhf(gates[gb + 1024]);
                const float oo = sigmoidf(gates[gb + 1536]);
                const size_t sb = (size_t)b2 * HID + j;
                const float cold = cbuf[sb];
                const float hold = hbuf[sb];
                const float cnew = ff * cold + ii * gg;
                const float ct   = u * cnew + um * cold;
                const float ht   = u * (oo * tanhf(ct)) + um * hold;
                cbuf[sb] = ct;
                hbuf[sb] = ht;
                hbf[sb]  = f2bf(ht);
                pd += ct * wu[j];
            }
            #pragma unroll
            for (int off = 16; off > 0; off >>= 1)
                pd += __shfl_xor(pd, off, 32);
            if (lane == 0) dotpart[b2 * 8 + ((e0 >> 6) & 7)] = pd;
        }
        gridBarrier(bar, gridDim.x);

        // ---- Phase 3: per-batch skip update gate ----
        if (blockIdx.x == 0 && threadIdx.x < B_SZ) {
            const int bb = threadIdx.x;
            float dot = 0.f;
            #pragma unroll
            for (int q = 0; q < 8; ++q) dot += dotpart[bb * 8 + q];
            const float du = sigmoidf(dot + bu[0]);
            const float ut = utbuf[bb];
            const float u  = ubin[bb];
            const float utn = u * du + (1.f - u) * (ut + fminf(du, 1.f - ut));
            utbuf[bb] = utn;
            ubin[bb]  = rintf(utn);          // round-half-even, matches jnp.round
        }
        gridBarrier(bar, gridDim.x);
    }

    // last hidden state -> output
    out[e0]     = hbuf[e0];
    out[e0 + 1] = hbuf[e0 + 1];
}

// ---------------- host launcher ----------------
extern "C" void kernel_launch(void* const* d_in, const int* in_sizes, int n_in,
                              void* d_out, int out_size, void* d_ws, size_t ws_size,
                              hipStream_t stream) {
    (void)in_sizes; (void)n_in; (void)out_size; (void)ws_size;
    const float* x      = (const float*)d_in[0];
    const float* conv_w = (const float*)d_in[1];
    const float* conv_b = (const float*)d_in[2];
    const float* gamma  = (const float*)d_in[3];
    const float* beta   = (const float*)d_in[4];
    const float* mean   = (const float*)d_in[5];
    const float* var    = (const float*)d_in[6];
    const float* w_ih   = (const float*)d_in[7];
    const float* w_hh   = (const float*)d_in[8];
    const float* b_ih   = (const float*)d_in[9];
    const float* b_hh   = (const float*)d_in[10];
    const float* w_u    = (const float*)d_in[11];
    const float* b_u    = (const float*)d_in[12];
    float* out = (float*)d_out;

    char* p = (char*)d_ws;
    auto alloc = [&](size_t bytes) {
        char* r = p;
        p += (bytes + 255) & ~(size_t)255;
        return r;
    };
    __bf16*   xbf     = (__bf16*)  alloc(2ull * B_SZ * T_SZ * D_INC);  // 16.8 MB
    __bf16*   seqbf   = (__bf16*)  alloc(2ull * B_SZ * L_SZ * NB);     // 16.7 MB
    __bf16*   wconv   = (__bf16*)  alloc(2ull * NB * KC);              // 0.33 MB
    __bf16*   wcomb   = (__bf16*)  alloc(2ull * G4 * KR);              // 3.1 MB
    float*    bcomb   = (float*)   alloc(4ull * G4);
    float*    bnscale = (float*)   alloc(4ull * NB);
    float*    bnshift = (float*)   alloc(4ull * NB);
    float*    gates   = (float*)   alloc(4ull * B_SZ * G4);            // 0.52 MB
    float*    hbuf    = (float*)   alloc(4ull * B_SZ * HID);
    float*    cbuf    = (float*)   alloc(4ull * B_SZ * HID);
    __bf16*   hbf     = (__bf16*)  alloc(2ull * B_SZ * HID);
    float*    utbuf   = (float*)   alloc(4ull * B_SZ);
    float*    ubin    = (float*)   alloc(4ull * B_SZ);
    float*    dotpart = (float*)   alloc(4ull * B_SZ * 8);
    unsigned* bar     = (unsigned*)alloc(256);

    prep_xbf<<<(B_SZ * T_SZ * D_INC + 255) / 256, 256, 0, stream>>>(x, xbf);
    prep_conv_w<<<(NB * KC + 255) / 256, 256, 0, stream>>>(conv_w, wconv);
    prep_comb_w<<<(G4 * KR + 255) / 256, 256, 0, stream>>>(w_ih, w_hh, wcomb);
    prep_misc<<<128, 256, 0, stream>>>(b_ih, b_hh, gamma, beta, mean, var,
                                       bcomb, bnscale, bnshift, hbuf, cbuf, hbf,
                                       utbuf, ubin, dotpart, bar);
    conv_bn_wmma<<<(B_SZ * L_SZ) / 16, 256, 0, stream>>>(xbf, wconv, conv_b,
                                                         bnscale, bnshift, seqbf);
    skiplstm_persistent<<<64, 256, SMEM_ELEMS * 2, stream>>>(
        seqbf, wcomb, bcomb, w_u, b_u, gates, hbuf, cbuf, hbf,
        utbuf, ubin, dotpart, bar, out);
}
